// PredictorPlus_8924942041235
// MI455X (gfx1250) — compile-verified, matrix operand order checked by reading, not run on hardware
//
#include <hip/hip_runtime.h>
#include <hip/hip_bf16.h>

typedef _Float16 f16;
typedef __attribute__((ext_vector_type(16))) _Float16 v16h;
typedef __attribute__((ext_vector_type(8))) float v8f;

#define ENTITY_SIZE 100000
#define NRQ 100          // rules per query (K of GEMM 1)
#define KPAD 128         // K padded to 4 x 32
#define HIDDEN 16
#define MLP_N 128
#define LN_EPS 1e-5f

// ---------------------------------------------------------------------------
// Pure-VALU 16-lane reduction via DPP16 (no ds_bpermute):
// quad_perm(xor1), quad_perm(xor2), row_half_mirror, row_mirror.
// Partial sums are uniform within each stage's group, so the mirrors supply
// the complementary group's value exactly like an xor-butterfly.
// ---------------------------------------------------------------------------
template <int CTRL>
__device__ __forceinline__ float dpp_add(float x) {
    int y = __builtin_amdgcn_update_dpp(0, __builtin_bit_cast(int, x),
                                        CTRL, 0xF, 0xF, true);
    return x + __builtin_bit_cast(float, y);
}
__device__ __forceinline__ float sum16(float x) {
    x = dpp_add<0xB1>(x);   // quad_perm [1,0,3,2]  (xor 1)
    x = dpp_add<0x4E>(x);   // quad_perm [2,3,0,1]  (xor 2)
    x = dpp_add<0x141>(x);  // row_half_mirror      (xor 4 over uniform quads)
    x = dpp_add<0x140>(x);  // row_mirror           (xor 8 over uniform octets)
    return x;
}

// ---------------------------------------------------------------------------
// Kernel 0: precompute re = rule_emb[rule_index] ([100][16]) and
// cvec[j] = b1[j] + sum_k rel[k] * W1[16+k][j]   ([128]) into workspace.
// ---------------------------------------------------------------------------
__global__ void pp_precompute(const int* __restrict__ all_r,
                              const int* __restrict__ rule_index,
                              const float* __restrict__ rule_emb,
                              const float* __restrict__ relation_emb,
                              const float* __restrict__ W1,
                              const float* __restrict__ b1,
                              float* __restrict__ ws_re,
                              float* __restrict__ ws_cvec) {
    int tid = threadIdx.x; // 128 threads
    if (tid < MLP_N) {
        int r = all_r[0];
        float acc = b1[tid];
#pragma unroll
        for (int k = 0; k < HIDDEN; ++k)
            acc += relation_emb[r * HIDDEN + k] * W1[(HIDDEN + k) * MLP_N + tid];
        ws_cvec[tid] = acc;
    }
    for (int i = tid; i < NRQ * HIDDEN; i += blockDim.x) {
        int ri = i >> 4, h = i & 15;
        ws_re[i] = rule_emb[rule_index[ri] * HIDDEN + h];
    }
}

// ---------------------------------------------------------------------------
// Kernel 1: init output with float4 stores: score = bias broadcast, mask = 1
// ---------------------------------------------------------------------------
__global__ void pp_init_out(float* __restrict__ out,
                            const float* __restrict__ bias,
                            int score_n, int total_n) {
    const int stride = gridDim.x * blockDim.x;
    const int nq = total_n >> 2;
    for (int q = blockIdx.x * blockDim.x + threadIdx.x; q < nq; q += stride) {
        const int i = q << 2;
        float4 v;
        v.x = (i + 0 < score_n) ? bias[(i + 0) % ENTITY_SIZE] : 1.0f;
        v.y = (i + 1 < score_n) ? bias[(i + 1) % ENTITY_SIZE] : 1.0f;
        v.z = (i + 2 < score_n) ? bias[(i + 2) % ENTITY_SIZE] : 1.0f;
        v.w = (i + 3 < score_n) ? bias[(i + 3) % ENTITY_SIZE] : 1.0f;
        reinterpret_cast<float4*>(out)[q] = v;
    }
    // tail (total_n not multiple of 4)
    const int base = total_n & ~3;
    const int t = blockIdx.x * blockDim.x + threadIdx.x;
    if (t < (total_n & 3))
        out[base + t] = (base + t < score_n) ? bias[(base + t) % ENTITY_SIZE] : 1.0f;
}

// ---------------------------------------------------------------------------
// Kernel 2: one wave per 16-candidate tile.
//  Stage 1: agg = rc_tile[16x100] (f16) @ re[100x16] via 4x wmma_f32_16x16x32_f16
//  Stage 2: LayerNorm+ReLU, DPP16 reductions (no DS)
//  Stage 3: h[16x16] @ W1a[16x128] via 8x wmma (K zero-padded to 32),
//           fused +cvec / ReLU / dot-W2
//  Stage 4: DPP reduce, scatter-store with bias
//
// f16 A-fragment layout (ISA 7.12.2, 16-bit A 16x32): lane half h owns
// K runs [8h, 8h+8) and [16+8h, 16+8h+8) per 32-K chunk; B mirrored.
// ---------------------------------------------------------------------------
__global__ __launch_bounds__(256) void pp_main(
    const float* __restrict__ rule_count,    // [100][NCAND]
    const int* __restrict__ candidate_set,   // [NCAND]
    const float* __restrict__ bias,          // [E]
    const float* __restrict__ ln_gamma,      // [16]
    const float* __restrict__ ln_beta,       // [16]
    const float* __restrict__ W1,            // [32][128]
    const float* __restrict__ W2,            // [128]
    const float* __restrict__ b2,            // [1]
    const float* __restrict__ ws_re,         // [100][16]
    const float* __restrict__ ws_cvec,       // [128]
    float* __restrict__ out,                 // score region
    int NCAND) {
    __shared__ f16   s_reT[HIDDEN * KPAD];   // [n][k], k zero-padded: 4 KB
    __shared__ f16   s_w1T[MLP_N * 32];      // [col][k], k>=16 zeroed: 8 KB
    __shared__ float s_cvec[MLP_N];          // 512 B
    __shared__ f16   s_h[8][16 * 32];        // per-wave [cand][k], k>=16 zero: 8 KB

    const int tid  = threadIdx.x;
    const int lane = tid & 31;
    const int wave = tid >> 5;
    const int l15  = lane & 15;
    const int h8   = (lane >> 4) << 3;       // 0 or 8: this half's K-run base

    // ---- cooperative staging (transpose + zero-pad + f32->f16) -----------
    for (int i = tid; i < HIDDEN * KPAD; i += 256) {
        int n = i >> 7, k = i & (KPAD - 1);
        s_reT[i] = (k < NRQ) ? (f16)ws_re[k * HIDDEN + n] : (f16)0.f;
    }
    for (int i = tid; i < MLP_N * 32; i += 256) {
        int col = i >> 5, k = i & 31;
        s_w1T[i] = (k < HIDDEN) ? (f16)W1[k * MLP_N + col] : (f16)0.f;
    }
    if (tid < MLP_N) s_cvec[tid] = ws_cvec[tid];
    for (int i = tid; i < 8 * 16 * 32; i += 256) (&s_h[0][0])[i] = (f16)0.f;
    __syncthreads();

    const int tile = blockIdx.x * 8 + wave;
    if (tile * 16 >= NCAND) return;          // wave-uniform
    const int cand = tile * 16 + l15;        // this lane's M row (column of rc)

    // ---- Stage 1: K=100 GEMM as 4 chained K=32 f16 WMMAs -----------------
    v8f acc = {0.f, 0.f, 0.f, 0.f, 0.f, 0.f, 0.f, 0.f};
#pragma unroll
    for (int kk = 0; kk < 4; ++kk) {
        v16h a, b;
#pragma unroll
        for (int j = 0; j < 16; ++j) {
            const int k = kk * 32 + ((j < 8) ? (h8 + j) : (16 + h8 + (j - 8)));
            // clamp row so the load is always legal (row stays cache-hot),
            // zero the lane via cndmask: keeps EXEC all-ones for WMMA.
            const int kc = (k < NRQ) ? k : (NRQ - 1);
            float v = rule_count[kc * NCAND + cand];
            a[j] = (f16)((k < NRQ) ? v : 0.f);
            b[j] = s_reT[l15 * KPAD + k];    // padded region already zero
        }
        acc = __builtin_amdgcn_wmma_f32_16x16x32_f16(
            false, a, false, b, (short)0, acc, false, false);
    }

    // ---- Stage 2: LayerNorm + ReLU (C/D layout: vgpr j = cand j+h8) ------
    const float g  = ln_gamma[l15];
    const float be = ln_beta[l15];
#pragma unroll
    for (int j = 0; j < 8; ++j) {
        const float x = acc[j];
        const float s  = sum16(x);
        const float s2 = sum16(x * x);
        const float mean = s * 0.0625f;
        const float var  = s2 * 0.0625f - mean * mean;
        float y = (x - mean) * rsqrtf(var + LN_EPS) * g + be;
        y = fmaxf(y, 0.f);
        s_h[wave][(j + h8) * 32 + l15] = (f16)y;   // LDS transpose, f16
    }

    // ---- Stage 3: [16x16] @ W1a via 8x K=32 WMMA (K zero-padded) ---------
    v16h ah;
#pragma unroll
    for (int j = 0; j < 16; ++j) {
        const int k = (j < 8) ? (h8 + j) : (16 + h8 + (j - 8));
        ah[j] = s_h[wave][l15 * 32 + k];     // k>=16 reads staged zeros
    }
    float outacc[8];
#pragma unroll
    for (int j = 0; j < 8; ++j) outacc[j] = 0.f;

#pragma unroll
    for (int nb = 0; nb < 8; ++nb) {
        v16h bw;
#pragma unroll
        for (int j = 0; j < 16; ++j) {
            const int k = (j < 8) ? (h8 + j) : (16 + h8 + (j - 8));
            bw[j] = s_w1T[(nb * 16 + l15) * 32 + k];
        }
        v8f acc2 = {0.f, 0.f, 0.f, 0.f, 0.f, 0.f, 0.f, 0.f};
        acc2 = __builtin_amdgcn_wmma_f32_16x16x32_f16(
            false, ah, false, bw, (short)0, acc2, false, false);
        const float cadd = s_cvec[nb * 16 + l15];
        const float w2v  = W2[nb * 16 + l15];
#pragma unroll
        for (int j = 0; j < 8; ++j)
            outacc[j] += fmaxf(acc2[j] + cadd, 0.f) * w2v;
    }

    // ---- Stage 4: DPP reduce over 16-lane groups, scatter ----------------
#pragma unroll
    for (int j = 0; j < 8; ++j) outacc[j] = sum16(outacc[j]);
    // lanes 0-7 own cands 0-7; lanes 16-23 own cands 8-15
    if (l15 < 8) {
        float my = outacc[0];
#pragma unroll
        for (int j = 1; j < 8; ++j) my = (l15 == j) ? outacc[j] : my;
        const int cl = l15 + h8;
        const int p  = candidate_set[tile * 16 + cl];   // < B*E, fits int32
        out[p] = my + b2[0] + bias[p % ENTITY_SIZE];
    }
}

// ---------------------------------------------------------------------------
extern "C" void kernel_launch(void* const* d_in, const int* in_sizes, int n_in,
                              void* d_out, int out_size, void* d_ws, size_t ws_size,
                              hipStream_t stream) {
    const int*   all_r         = (const int*)d_in[1];
    const int*   candidate_set = (const int*)d_in[2];
    const int*   rule_index    = (const int*)d_in[3];
    const float* rule_count    = (const float*)d_in[4];
    const float* rule_emb      = (const float*)d_in[5];
    const float* relation_emb  = (const float*)d_in[6];
    const float* bias          = (const float*)d_in[7];
    const float* ln_gamma      = (const float*)d_in[8];
    const float* ln_beta       = (const float*)d_in[9];
    const float* W1            = (const float*)d_in[10];
    const float* b1            = (const float*)d_in[11];
    const float* W2            = (const float*)d_in[12];
    const float* b2            = (const float*)d_in[13];

    const int B       = in_sizes[0];         // 256
    const int NCAND   = in_sizes[2];         // 1,000,000
    const int score_n = B * ENTITY_SIZE;     // 25.6M

    float* ws_re   = (float*)d_ws;                 // 100*16 floats
    float* ws_cvec = ws_re + NRQ * HIDDEN;         // 128 floats
    float* outp    = (float*)d_out;

    // 0) gather rule embeddings + constant MLP vector
    pp_precompute<<<1, 128, 0, stream>>>(all_r, rule_index, rule_emb,
                                         relation_emb, W1, b1, ws_re, ws_cvec);

    // 1) init score = bias broadcast, mask = 1.0 (float4 stores)
    {
        int blocks = ((out_size >> 2) + 255) / 256;
        if (blocks > 65536) blocks = 65536;
        if (blocks < 1) blocks = 1;
        pp_init_out<<<blocks, 256, 0, stream>>>(outp, bias, score_n, out_size);
    }

    // 2) fused GEMM -> LN -> MLP -> scatter
    {
        const int ntiles = (NCAND + 15) / 16;        // 62500
        const int blocks = (ntiles + 7) / 8;         // 8 waves / block
        pp_main<<<blocks, 256, 0, stream>>>(rule_count, candidate_set, bias,
                                            ln_gamma, ln_beta, W1, W2, b2,
                                            ws_re, ws_cvec, outp, NCAND);
    }
}
